// EncoderLayer_24824910971449
// MI455X (gfx1250) — compile-verified, matrix operand order checked by reading, not run on hardware
//
#include <hip/hip_runtime.h>
#include <hip/hip_bf16.h>

typedef __bf16 bf16_t;
typedef __attribute__((ext_vector_type(16))) __bf16 v16bf;
typedef __attribute__((ext_vector_type(8)))  float  v8f;
typedef unsigned int u32x4 __attribute__((ext_vector_type(4)));
typedef int          i32x4 __attribute__((ext_vector_type(4)));
typedef int          i32x8 __attribute__((ext_vector_type(8)));

union FragA { v16bf v; uint4 u[2]; };
union F16x8 { uint4 u; bf16_t h[8]; };

#if __has_builtin(__builtin_amdgcn_tensor_load_to_lds) && __has_builtin(__builtin_amdgcn_s_wait_tensorcnt)
#define HAS_TDM 1
#define WAIT_TENSOR(n) __builtin_amdgcn_s_wait_tensorcnt(n)
#else
#define HAS_TDM 0
#define WAIT_TENSOR(n)
#endif

// ---------------------------------------------------------------------------
// TDM: 2D tile (tile_w elems of 2B) x tile_h rows, global row stride in elems,
// optional LDS padding (pad_amount dwords after every (1<<pad_interval)*8 bytes).
// D# packing per CDNA5 ISA ch.8 (group0 128b: count/lds_addr/global_addr/type,
// group1 256b: data_size/dims/tiles/strides). Groups 2/3 zero (2D tensor).
// ---------------------------------------------------------------------------
#if HAS_TDM
__device__ __forceinline__ void tdm_load_2d(
    unsigned lds_off, const void* gptr,
    unsigned tile_w, unsigned tile_h, unsigned long long row_stride_elems,
    unsigned pad_interval_code, unsigned pad_amount_dwords /*0 = no pad*/)
{
    unsigned long long ga = (unsigned long long)(uintptr_t)gptr;
    u32x4 g0;
    g0[0] = 1u;                                              // count=1, user mode
    g0[1] = lds_off;                                         // lds_addr
    g0[2] = (unsigned)ga;                                    // global_addr[31:0]
    g0[3] = (unsigned)((ga >> 32) & 0x01FFFFFFu) | 0x80000000u; // addr[56:32] | type=2
    unsigned w0 = 0x00010000u;                               // data_size=1 (2 bytes)
    if (pad_amount_dwords) {
        w0 |= (1u << 20);                                    // pad_enable
        w0 |= (pad_interval_code & 7u) << 22;
        w0 |= ((pad_amount_dwords - 1u) & 127u) << 25;
    }
    i32x8 g1;
    g1[0] = (int)w0;
    g1[1] = (int)((tile_w & 0xFFFFu) << 16);                 // tensor_dim0[15:0] = tile_w
    g1[2] = (int)((tile_h & 0xFFFFu) << 16);                 // tensor_dim1[15:0] = tile_h
    g1[3] = (int)((tile_w & 0xFFFFu) << 16);                 // tile_dim0 = tile_w
    g1[4] = (int)(tile_h & 0xFFFFu);                         // tile_dim1 (tile_dim2=0)
    g1[5] = (int)(unsigned)(row_stride_elems & 0xFFFFFFFFull); // dim0_stride[31:0]
    g1[6] = (int)(unsigned)((row_stride_elems >> 32) & 0xFFFFull); // dim0_stride[47:32]
    g1[7] = 0;
    i32x4 z4 = {0, 0, 0, 0};
#if __clang_major__ >= 23
    i32x8 z8 = {0, 0, 0, 0, 0, 0, 0, 0};
    __builtin_amdgcn_tensor_load_to_lds(g0, g1, z4, z4, z8, 0);
#else
    __builtin_amdgcn_tensor_load_to_lds(g0, g1, z4, z4, 0);
#endif
}
#endif

// ---------------------------------------------------------------------------
// 16x32 bf16 A/B fragment from a row-major LDS tile (ISA 7.12.2 layout):
// lane L -> row L&15, kbase=(L>>4)*8, elems = K[kb..kb+7], K[kb+16..kb+23].
// ---------------------------------------------------------------------------
__device__ __forceinline__ v16bf load_frag(const bf16_t* base, int stride, int lane) {
    int row = lane & 15;
    int kb  = (lane >> 4) * 8;
    const bf16_t* p = base + row * stride + kb;
    FragA f;
    f.u[0] = *(const uint4*)(p);
    f.u[1] = *(const uint4*)(p + 16);
    return f.v;
}

__device__ __forceinline__ v8f wmma_bf16(v16bf a, v16bf b, v8f c) {
    return __builtin_amdgcn_wmma_f32_16x16x32_bf16(false, a, false, b, (short)0, c, false, false);
}

// ---------------------------------------------------------------------------
// f32 -> bf16 conversion
// ---------------------------------------------------------------------------
__global__ void cvt_bf16_kernel(const float* __restrict__ in, bf16_t* __restrict__ out, int n) {
    int i = blockIdx.x * blockDim.x + threadIdx.x;
    if (i < n) out[i] = (bf16_t)in[i];
}

// ---------------------------------------------------------------------------
// C[M,N] = A[M,K](bf16 rm) @ W[N,K](bf16 rm)^T + bias[N]
// 128x128 block tile, 256 thr = 8 wave32 (4x2), wave tile 32x64.
// TDM double-buffered K staging (2 tensor ops in flight), WMMA inner loop.
// ---------------------------------------------------------------------------
template <bool OUT_BF16, bool RELU>
__global__ __launch_bounds__(256) void gemm_wmma_kernel(
    const bf16_t* __restrict__ A, const bf16_t* __restrict__ W,
    const float* __restrict__ bias, void* __restrict__ Cout,
    int M, int N, int K)
{
    __shared__ bf16_t As[2][128 * 32];
    __shared__ bf16_t Ws[2][128 * 32];

    const int tid  = threadIdx.x;
    const int lane = tid & 31;
    const int wid  = tid >> 5;
    const int wm   = wid & 3;
    const int wn   = wid >> 2;
    const int bm   = blockIdx.x * 128;
    const int bn   = blockIdx.y * 128;
    const int T    = K >> 5;

    v8f acc[2][4];
    #pragma unroll
    for (int i = 0; i < 2; ++i)
        #pragma unroll
        for (int j = 0; j < 4; ++j) acc[i][j] = (v8f){0.f,0.f,0.f,0.f,0.f,0.f,0.f,0.f};

#if HAS_TDM
    if (tid == 0) {
        tdm_load_2d((unsigned)(uintptr_t)&As[0][0], &A[(size_t)bm * K], 32, 128, (unsigned)K, 0, 0);
        tdm_load_2d((unsigned)(uintptr_t)&Ws[0][0], &W[(size_t)bn * K], 32, 128, (unsigned)K, 0, 0);
    }
#endif

    for (int it = 0; it < T; ++it) {
        const int buf = it & 1;
#if HAS_TDM
        if (it + 1 < T) {
            if (tid == 0) {
                int kk = (it + 1) << 5;
                tdm_load_2d((unsigned)(uintptr_t)&As[buf ^ 1][0], &A[(size_t)bm * K + kk], 32, 128, (unsigned)K, 0, 0);
                tdm_load_2d((unsigned)(uintptr_t)&Ws[buf ^ 1][0], &W[(size_t)bn * K + kk], 32, 128, (unsigned)K, 0, 0);
            }
            WAIT_TENSOR(2);          // current tile's 2 in-order TDM ops complete
        } else {
            WAIT_TENSOR(0);
        }
        __syncthreads();             // publish TDM completion to all waves
#else
        const int k0 = it << 5;
        __syncthreads();
        #pragma unroll
        for (int c = tid; c < 512; c += 256) {
            int row = c >> 2, col = (c & 3) * 8;
            *(uint4*)&As[buf][row * 32 + col] = *(const uint4*)&A[(size_t)(bm + row) * K + k0 + col];
        }
        #pragma unroll
        for (int c = tid; c < 512; c += 256) {
            int row = c >> 2, col = (c & 3) * 8;
            *(uint4*)&Ws[buf][row * 32 + col] = *(const uint4*)&W[(size_t)(bn + row) * K + k0 + col];
        }
        __syncthreads();
#endif
        v16bf af[2], wf[4];
        #pragma unroll
        for (int i = 0; i < 2; ++i)
            af[i] = load_frag(&As[buf][(wm * 32 + i * 16) * 32], 32, lane);
        #pragma unroll
        for (int j = 0; j < 4; ++j)
            wf[j] = load_frag(&Ws[buf][(wn * 64 + j * 16) * 32], 32, lane);
        #pragma unroll
        for (int i = 0; i < 2; ++i)
            #pragma unroll
            for (int j = 0; j < 4; ++j)
                acc[i][j] = wmma_bf16(af[i], wf[j], acc[i][j]);
        __syncthreads();             // reads done before buffer is overwritten
    }

    const int g  = lane >> 4;
    const int ln = lane & 15;
    #pragma unroll
    for (int i = 0; i < 2; ++i) {
        #pragma unroll
        for (int j = 0; j < 4; ++j) {
            int col = bn + wn * 64 + j * 16 + ln;
            float bv = bias[col];
            #pragma unroll
            for (int r = 0; r < 8; ++r) {
                int row = bm + wm * 32 + i * 16 + r + 8 * g;
                float val = acc[i][j][r] + bv;
                if (RELU) val = fmaxf(val, 0.f);
                if (OUT_BF16) ((bf16_t*)Cout)[(size_t)row * N + col] = (bf16_t)val;
                else          ((float*)Cout)[(size_t)row * N + col]  = val;
            }
        }
    }
}

// ---------------------------------------------------------------------------
// Flash attention: q,k,v bf16 [B,S,H*hd]; one (b,h) + 64 q rows per block.
// 128 thr = 4 waves; online softmax over 32-key tiles; hd = 64.
// K tile staged by TDM with LDS padding (stride 72); V transposed manually.
// ---------------------------------------------------------------------------
__global__ __launch_bounds__(128) void flash_attn_kernel(
    const bf16_t* __restrict__ q, const bf16_t* __restrict__ k,
    const bf16_t* __restrict__ v, bf16_t* __restrict__ ctx)
{
    const int S = 2048, DM = 1024;
    __shared__ bf16_t Ks[32 * 72];      // [key][d], stride 72 (TDM pad 4dw/32dw)
    __shared__ bf16_t Vt[64 * 40];      // [d][key] transposed, padded stride
    __shared__ bf16_t Ps[4][16 * 32];   // per-wave P staging (C->A relayout)

    const int tid  = threadIdx.x;
    const int lane = tid & 31;
    const int w    = tid >> 5;
    const int qt = blockIdx.x & 31;
    const int h  = (blockIdx.x >> 5) & 15;
    const int b  = blockIdx.x >> 9;
    const size_t base = (size_t)b * S * DM + (size_t)h * 64;
    const int qrow0 = qt * 64 + w * 16;

    // q fragments (16 rows x 64 -> two K=32 chunks), 1/sqrt(64) folded in.
    v16bf qa[2];
    {
        int row = lane & 15, kb = (lane >> 4) * 8;
        #pragma unroll
        for (int c = 0; c < 2; ++c) {
            const bf16_t* p = q + base + (size_t)(qrow0 + row) * DM + c * 32 + kb;
            FragA f;
            f.u[0] = *(const uint4*)(p);
            f.u[1] = *(const uint4*)(p + 16);
            #pragma unroll
            for (int i = 0; i < 16; ++i) f.v[i] = (bf16_t)((float)f.v[i] * 0.125f);
            qa[c] = f.v;
        }
    }

    float m_i[8], l_i[8];
    v8f o[4];
    #pragma unroll
    for (int r = 0; r < 8; ++r) { m_i[r] = -1e30f; l_i[r] = 0.f; }
    #pragma unroll
    for (int j = 0; j < 4; ++j) o[j] = (v8f){0.f,0.f,0.f,0.f,0.f,0.f,0.f,0.f};

    const int g  = lane >> 4;
    const int ln = lane & 15;

    for (int kt = 0; kt < S; kt += 32) {
        __syncthreads();                        // prior reads of Ks/Vt/Ps done
#if HAS_TDM
        if (tid == 0)                           // 32 keys x 64 elems, pad 4dw per 32dw
            tdm_load_2d((unsigned)(uintptr_t)&Ks[0], &k[base + (size_t)kt * DM],
                        64, 32, (unsigned)DM, 4, 4);
#else
        #pragma unroll
        for (int c = tid; c < 256; c += 128) {
            int key = c >> 3, col = (c & 7) * 8;
            *(uint4*)&Ks[key * 72 + col] =
                *(const uint4*)&k[base + (size_t)(kt + key) * DM + col];
        }
#endif
        // V tile, transposed into [d][key].
        #pragma unroll
        for (int c = tid; c < 256; c += 128) {
            int key = c >> 3, col = (c & 7) * 8;
            F16x8 t;
            t.u = *(const uint4*)&v[base + (size_t)(kt + key) * DM + col];
            #pragma unroll
            for (int i = 0; i < 8; ++i) Vt[(col + i) * 40 + key] = t.h[i];
        }
        if (kt + 32 < S)
            __builtin_prefetch(&v[base + (size_t)(kt + 32 + (tid >> 2)) * DM], 0, 1);
        WAIT_TENSOR(0);
        __syncthreads();

        // scores: all 4 K-fragments first, then 4 WMMAs (overlap DS latency).
        v16bf kf[2][2];
        #pragma unroll
        for (int c = 0; c < 2; ++c) {
            kf[c][0] = load_frag(&Ks[c * 32], 72, lane);
            kf[c][1] = load_frag(&Ks[16 * 72 + c * 32], 72, lane);
        }
        v8f s0 = (v8f){0.f,0.f,0.f,0.f,0.f,0.f,0.f,0.f};
        v8f s1 = s0;
        s0 = wmma_bf16(qa[0], kf[0][0], s0);
        s1 = wmma_bf16(qa[0], kf[0][1], s1);
        s0 = wmma_bf16(qa[1], kf[1][0], s0);
        s1 = wmma_bf16(qa[1], kf[1][1], s1);

        // online softmax per row (row = r + 8*g, one 16-lane half).
        #pragma unroll
        for (int r = 0; r < 8; ++r) {
            float tm = fmaxf(s0[r], s1[r]);
            #pragma unroll
            for (int msk = 8; msk >= 1; msk >>= 1) tm = fmaxf(tm, __shfl_xor(tm, msk));
            float mn   = fmaxf(m_i[r], tm);
            float corr = __expf(m_i[r] - mn);
            float p0 = __expf(s0[r] - mn);
            float p1 = __expf(s1[r] - mn);
            float rs = p0 + p1;
            #pragma unroll
            for (int msk = 8; msk >= 1; msk >>= 1) rs += __shfl_xor(rs, msk);
            l_i[r] = l_i[r] * corr + rs;
            m_i[r] = mn;
            #pragma unroll
            for (int j = 0; j < 4; ++j) o[j][r] *= corr;
            s0[r] = p0;
            s1[r] = p1;
        }

        // P: C layout -> LDS 16x32 -> reload as A fragment.
        #pragma unroll
        for (int r = 0; r < 8; ++r) {
            int row = r + 8 * g;
            Ps[w][row * 32 + ln]      = (bf16_t)s0[r];
            Ps[w][row * 32 + ln + 16] = (bf16_t)s1[r];
        }
        __syncthreads();

        v16bf pa = load_frag(&Ps[w][0], 32, lane);
        v16bf vb[4];
        #pragma unroll
        for (int j = 0; j < 4; ++j) vb[j] = load_frag(&Vt[(j * 16) * 40], 40, lane);
        #pragma unroll
        for (int j = 0; j < 4; ++j) o[j] = wmma_bf16(pa, vb[j], o[j]);
    }

    float inv[8];
    #pragma unroll
    for (int r = 0; r < 8; ++r) inv[r] = 1.f / l_i[r];
    #pragma unroll
    for (int j = 0; j < 4; ++j) {
        #pragma unroll
        for (int r = 0; r < 8; ++r) {
            int row = qrow0 + r + 8 * g;
            ctx[base + (size_t)row * DM + j * 16 + ln] = (bf16_t)(o[j][r] * inv[r]);
        }
    }
}

// ---------------------------------------------------------------------------
// out = LayerNorm(A + B) * gamma + beta ; one block (256 thr) per row, D=1024.
// ---------------------------------------------------------------------------
template <bool OUT_BF16>
__global__ __launch_bounds__(256) void add_ln_kernel(
    const float* __restrict__ A, const float* __restrict__ B,
    const float* __restrict__ gam, const float* __restrict__ bet,
    void* __restrict__ out)
{
    const int D = 1024;
    const int row = blockIdx.x, tid = threadIdx.x;
    const int lane = tid & 31, wid = tid >> 5;
    __shared__ float red1[8], red2[8];

    const size_t rb = (size_t)row * D;
    float x[4];
    float s = 0.f;
    #pragma unroll
    for (int i = 0; i < 4; ++i) {
        int c = tid * 4 + i;
        x[i] = A[rb + c] + B[rb + c];
        s += x[i];
    }
    #pragma unroll
    for (int msk = 16; msk >= 1; msk >>= 1) s += __shfl_xor(s, msk);
    if (lane == 0) red1[wid] = s;
    __syncthreads();
    float tot = 0.f;
    #pragma unroll
    for (int i = 0; i < 8; ++i) tot += red1[i];
    const float mean = tot * (1.f / 1024.f);

    float ss = 0.f;
    #pragma unroll
    for (int i = 0; i < 4; ++i) { float d = x[i] - mean; ss += d * d; }
    #pragma unroll
    for (int msk = 16; msk >= 1; msk >>= 1) ss += __shfl_xor(ss, msk);
    if (lane == 0) red2[wid] = ss;
    __syncthreads();
    float vtot = 0.f;
    #pragma unroll
    for (int i = 0; i < 8; ++i) vtot += red2[i];
    const float rstd = rsqrtf(vtot * (1.f / 1024.f) + 1e-5f);

    #pragma unroll
    for (int i = 0; i < 4; ++i) {
        int c = tid * 4 + i;
        float val = (x[i] - mean) * rstd * gam[c] + bet[c];
        if (OUT_BF16) ((bf16_t*)out)[rb + c] = (bf16_t)val;
        else          ((float*)out)[rb + c]  = val;
    }
}

// ---------------------------------------------------------------------------
// Host orchestration.
// ---------------------------------------------------------------------------
extern "C" void kernel_launch(void* const* d_in, const int* in_sizes, int n_in,
                              void* d_out, int out_size, void* d_ws, size_t ws_size,
                              hipStream_t stream) {
    const int Mi = 8192, Di = 1024, Fi = 4096;

    const float* x  = (const float*)d_in[0];
    const float* Wq = (const float*)d_in[1];  const float* bq = (const float*)d_in[2];
    const float* Wk = (const float*)d_in[3];  const float* bk = (const float*)d_in[4];
    const float* Wv = (const float*)d_in[5];  const float* bv = (const float*)d_in[6];
    const float* Wo = (const float*)d_in[7];  const float* bo = (const float*)d_in[8];
    const float* W1 = (const float*)d_in[9];  const float* b1 = (const float*)d_in[10];
    const float* W2 = (const float*)d_in[11]; const float* b2 = (const float*)d_in[12];
    const float* g1 = (const float*)d_in[13]; const float* be1 = (const float*)d_in[14];
    const float* g2 = (const float*)d_in[15]; const float* be2 = (const float*)d_in[16];

    char* wp = (char*)d_ws;
    auto take = [&](size_t bytes) -> void* {
        void* r = (void*)wp;
        wp += (bytes + 255) & ~(size_t)255;
        return r;
    };
    const size_t MD = (size_t)Mi * Di, DD = (size_t)Di * Di, FD = (size_t)Fi * Di, MF = (size_t)Mi * Fi;
    bf16_t* xb   = (bf16_t*)take(MD * 2);
    bf16_t* Wqb  = (bf16_t*)take(DD * 2);
    bf16_t* Wkb  = (bf16_t*)take(DD * 2);
    bf16_t* Wvb  = (bf16_t*)take(DD * 2);
    bf16_t* Wob  = (bf16_t*)take(DD * 2);
    bf16_t* W1b  = (bf16_t*)take(FD * 2);
    bf16_t* W2b  = (bf16_t*)take(FD * 2);
    bf16_t* qb   = (bf16_t*)take(MD * 2);
    bf16_t* kb_  = (bf16_t*)take(MD * 2);
    bf16_t* vb_  = (bf16_t*)take(MD * 2);
    bf16_t* ctxb = (bf16_t*)take(MD * 2);
    float*  out1 = (float*)take(MD * 4);
    bf16_t* ln1b = (bf16_t*)take(MD * 2);
    bf16_t* midb = (bf16_t*)take(MF * 2);
    float*  out2 = (float*)take(MD * 4);

    auto cvt = [&](const float* src, bf16_t* dst, size_t n) {
        cvt_bf16_kernel<<<(unsigned)((n + 255) / 256), 256, 0, stream>>>(src, dst, (int)n);
    };
    cvt(x,  xb,  MD);
    cvt(Wq, Wqb, DD); cvt(Wk, Wkb, DD); cvt(Wv, Wvb, DD); cvt(Wo, Wob, DD);
    cvt(W1, W1b, FD); cvt(W2, W2b, FD);

    dim3 gDD(Mi / 128, Di / 128);
    gemm_wmma_kernel<true, false><<<gDD, 256, 0, stream>>>(xb, Wqb, bq, qb,  Mi, Di, Di);
    gemm_wmma_kernel<true, false><<<gDD, 256, 0, stream>>>(xb, Wkb, bk, kb_, Mi, Di, Di);
    gemm_wmma_kernel<true, false><<<gDD, 256, 0, stream>>>(xb, Wvb, bv, vb_, Mi, Di, Di);

    flash_attn_kernel<<<2048, 128, 0, stream>>>(qb, kb_, vb_, ctxb);

    gemm_wmma_kernel<false, false><<<gDD, 256, 0, stream>>>(ctxb, Wob, bo, out1, Mi, Di, Di);
    add_ln_kernel<true><<<Mi, 256, 0, stream>>>(x, out1, g1, be1, ln1b);

    dim3 gDF(Mi / 128, Fi / 128);
    gemm_wmma_kernel<true, true><<<gDF, 256, 0, stream>>>(ln1b, W1b, b1, midb, Mi, Fi, Di);
    gemm_wmma_kernel<false, false><<<gDD, 256, 0, stream>>>(midb, W2b, b2, out2, Mi, Di, Fi);

    add_ln_kernel<false><<<Mi, 256, 0, stream>>>(out1, out2, g2, be2, d_out);
}
